// XDecoder_17051020165544
// MI455X (gfx1250) — matrix operand order repaired and checked by reading, not compiled, not run on hardware
//
#include <hip/hip_runtime.h>
#include <hip/hip_bf16.h>
#include <math.h>

// ---------------------------------------------------------------------------
// MI455X (gfx1250) implementation.
// ~420 GFLOP of GEMM-dominated work; weights+activations fit in the 192MB L2
// -> compute bound on the matrix pipes. All large GEMMs and both attention
// matmuls run on v_wmma_f32_16x16x32_bf16 (bf16 in / f32 accumulate).
// GEMM is software-pipelined with double-buffered LDS: global loads for
// tile t+1 are issued before the WMMAs of tile t, so s_wait_loadcnt lands
// after the matrix work instead of after every load.
// ---------------------------------------------------------------------------

typedef __bf16  bf16x16 __attribute__((ext_vector_type(16)));
typedef __bf16  bf16x8  __attribute__((ext_vector_type(8)));
typedef __bf16  bf16x4  __attribute__((ext_vector_type(4)));
typedef float   f32x8   __attribute__((ext_vector_type(8)));

#define NB 8
#define NS 1024
#define NDIM 512
#define NHEADS 8
#define NDH 64
#define NDEPTH 6
#define NFF 2048
#define NDIN 524
#define NEMB 512
#define NNCLS 1024

#define TM 128
#define TN 64
#define TK 32
#define ASTRIDE 40   // 32 + 8 pad  (80B rows: 16B aligned, bank-spread)
#define BSTRIDE 40
#define QSTRIDE 72   // 64 + 8 pad  (144B rows)
#define KSTRIDE 72

// A-fragment permutation: element order of the 16-bit 16x32 A layout
// (ISA 7.12.2). For kk in [0,32): lanes want [half*8..+7, 16+half*8..+7]
// contiguous -> p(kk) = ((kk>>3)&1)*16 + ((kk>>4)&1)*8 + (kk&7).
__device__ __forceinline__ int apos(int kk) {
    return (((kk >> 3) & 1) << 4) + (((kk >> 4) & 1) << 3) + (kk & 7);
}

__device__ __forceinline__ bf16x16 ld_frag(const __bf16* p) {
    bf16x8 lo = *(const bf16x8*)p;
    bf16x8 hi = *(const bf16x8*)(p + 8);
    return __builtin_shufflevector(lo, hi, 0, 1, 2, 3, 4, 5, 6, 7,
                                   8, 9, 10, 11, 12, 13, 14, 15);
}

__device__ __forceinline__ void st4(__bf16* p, float a, float b, float c, float d) {
    bf16x4 v = { (__bf16)a, (__bf16)b, (__bf16)c, (__bf16)d };
    *(bf16x4*)p = v;
}

#define WMMA_BF16(A, Bf, C) \
    __builtin_amdgcn_wmma_f32_16x16x32_bf16(false, A, false, Bf, (short)0, C, false, false)

// ---------------------------------------------------------------------------
// GEMM: C[M,N] = epilogue( A[M,lda](:,0:K) @ W[K,N] ); +bias / GELU / +res.
// 128x64x32 macro-tile, 8 waves; each wave owns 32x32 = 4 WMMAs per K-step.
// Double-buffered LDS + register prefetch pipeline.
// ---------------------------------------------------------------------------
__global__ __launch_bounds__(256) void gemm_bf16_wmma(
    const float* __restrict__ A, const float* __restrict__ W,
    const float* __restrict__ bias, const float* __restrict__ residual,
    float* __restrict__ C, int N, int K, int lda, int act)
{
    __shared__ __align__(16) __bf16 As2[2][TM * ASTRIDE];   // natural [m][k] order
    __shared__ __align__(16) __bf16 Bs2[2][TN * BSTRIDE];   // transposed: [n][k]

    const int tid  = threadIdx.x;
    const int lane = tid & 31;
    const int wave = tid >> 5;
    const int wm   = wave >> 1;       // 0..3 : 32-row strip
    const int wn   = wave & 1;        // 0..1 : 32-col strip
    const int half = lane >> 4;
    const int l16  = lane & 15;

    const int block_m = blockIdx.x * TM;
    const int block_n = blockIdx.y * TN;

    f32x8 acc00 = {}, acc01 = {}, acc10 = {}, acc11 = {};

    float4 areg[4];
    float  breg[2][4];

    // --- register prefetch: pure loads, no dependent ALU in between
    auto load_tile = [&](int k0) {
        #pragma unroll
        for (int i = 0; i < 4; ++i) {
            int idx4 = tid + i * 256;             // 0..1023
            int r  = idx4 >> 3;                   // 8 float4 per 32-wide row
            int c4 = (idx4 & 7) * 4;
            areg[i] = *(const float4*)(A + (size_t)(block_m + r) * lda + k0 + c4);
        }
        #pragma unroll
        for (int i = 0; i < 2; ++i) {
            int t   = tid + i * 256;              // 0..511
            int col = t & 63;                     // coalesced over columns
            int r4  = (t >> 6) << 2;              // 0,4,...,28
            int gcol = block_n + col;
            int cc   = gcol < N ? gcol : N - 1;   // clamp (branchless)
            const float* wp = W + (size_t)(k0 + r4) * N + cc;
            breg[i][0] = wp[0];
            breg[i][1] = wp[(size_t)N];
            breg[i][2] = wp[2 * (size_t)N];
            breg[i][3] = wp[3 * (size_t)N];
        }
    };
    // --- convert + packed b64 LDS stores (mask applied here, not at load)
    auto store_tile = [&](int buf) {
        #pragma unroll
        for (int i = 0; i < 4; ++i) {
            int idx4 = tid + i * 256;
            int r  = idx4 >> 3;
            int c4 = (idx4 & 7) * 4;
            st4(&As2[buf][r * ASTRIDE + c4],
                areg[i].x, areg[i].y, areg[i].z, areg[i].w);
        }
        #pragma unroll
        for (int i = 0; i < 2; ++i) {
            int t   = tid + i * 256;
            int col = t & 63;
            int r4  = (t >> 6) << 2;
            float m = (block_n + col) < N ? 1.f : 0.f;
            st4(&Bs2[buf][col * BSTRIDE + r4],
                breg[i][0] * m, breg[i][1] * m, breg[i][2] * m, breg[i][3] * m);
        }
    };

    const int nk = K / TK;
    load_tile(0);
    store_tile(0);
    __syncthreads();

    for (int t = 0; t < nk; ++t) {
        const int buf = t & 1;
        if (t + 1 < nk) load_tile((t + 1) * TK);   // issue next-tile loads early

        bf16x16 af0 = ld_frag(&As2[buf][(wm * 32 + l16) * ASTRIDE + half * 16]);
        bf16x16 af1 = ld_frag(&As2[buf][(wm * 32 + 16 + l16) * ASTRIDE + half * 16]);
        bf16x16 bf0 = ld_frag(&Bs2[buf][(wn * 32 + l16) * BSTRIDE + half * 16]);
        bf16x16 bf1 = ld_frag(&Bs2[buf][(wn * 32 + 16 + l16) * BSTRIDE + half * 16]);

        acc00 = WMMA_BF16(af0, bf0, acc00);
        acc01 = WMMA_BF16(af0, bf1, acc01);
        acc10 = WMMA_BF16(af1, bf0, acc10);
        acc11 = WMMA_BF16(af1, bf1, acc11);

        if (t + 1 < nk) store_tile(buf ^ 1);       // waits for loads land here
        __syncthreads();
    }

    // ---- epilogue: C frag row = r + 8*half, col = l16
    #pragma unroll
    for (int mi = 0; mi < 2; ++mi) {
        #pragma unroll
        for (int ni = 0; ni < 2; ++ni) {
            #pragma unroll
            for (int r = 0; r < 8; ++r) {
                int row = block_m + wm * 32 + mi * 16 + half * 8 + r;
                int col = block_n + wn * 32 + ni * 16 + l16;
                if (col < N) {
                    float v = (mi == 0) ? ((ni == 0) ? acc00[r] : acc01[r])
                                        : ((ni == 0) ? acc10[r] : acc11[r]);
                    if (bias)     v += bias[col];
                    if (act == 1) v = 0.5f * v * (1.0f + erff(v * 0.70710678118f));
                    if (residual) v += residual[(size_t)row * N + col];
                    C[(size_t)row * N + col] = v;
                }
            }
        }
    }
}

// ---------------------------------------------------------------------------
// Embedding gather + concat + proj_in (K=524, irregular -> VALU; ~1% of FLOPs)
// ---------------------------------------------------------------------------
__global__ __launch_bounds__(256) void embed_proj_in_k(
    const float* __restrict__ saml, const float* __restrict__ emb,
    const float* __restrict__ Wp, const float* __restrict__ bp,
    float* __restrict__ h)
{
    int row = blockIdx.x;                  // 0..B*S-1
    int b = row >> 10, s = row & 1023;
    __shared__ float xrow[NDIN];
    const float* src = saml + ((size_t)b * 1025 + s) * 13;
    int idx = (int)src[0];
    for (int i = threadIdx.x; i < NEMB; i += 256)
        xrow[i] = emb[(size_t)idx * NEMB + i];
    if (threadIdx.x < 12) xrow[NEMB + threadIdx.x] = src[1 + threadIdx.x];
    __syncthreads();
    for (int n = threadIdx.x; n < NDIM; n += 256) {
        float acc = bp[n];
        for (int kk = 0; kk < NDIN; ++kk)
            acc = fmaf(xrow[kk], Wp[(size_t)kk * NDIM + n], acc);
        h[(size_t)row * NDIM + n] = acc;
    }
}

// ---------------------------------------------------------------------------
// ScaleNorm: y = x / max(||x|| * DIM^-0.5, 1e-5) * g[l]
// ---------------------------------------------------------------------------
__global__ __launch_bounds__(256) void scalenorm_k(
    const float* __restrict__ x, float* __restrict__ y,
    const float* __restrict__ gv, int gidx)
{
    int row = blockIdx.x;
    const float* xr = x + (size_t)row * NDIM;
    __shared__ float red[8];
    float ss = 0.f;
    for (int i = threadIdx.x; i < NDIM; i += 256) { float v = xr[i]; ss += v * v; }
    #pragma unroll
    for (int o = 16; o > 0; o >>= 1) ss += __shfl_xor(ss, o, 32);
    if ((threadIdx.x & 31) == 0) red[threadIdx.x >> 5] = ss;
    __syncthreads();
    if (threadIdx.x == 0) {
        float t = 0.f;
        #pragma unroll
        for (int i = 0; i < 8; ++i) t += red[i];
        red[0] = t;
    }
    __syncthreads();
    float n = sqrtf(red[0]) * 0.04419417382f;   // * DIM^-0.5
    float scale = gv[gidx] / fmaxf(n, 1e-5f);
    for (int i = threadIdx.x; i < NDIM; i += 256)
        y[(size_t)row * NDIM + i] = xr[i] * scale;
}

// ---------------------------------------------------------------------------
// Rotary on first 32 dims of each head (pairs t, t+16), applied to q and k.
// ---------------------------------------------------------------------------
__global__ __launch_bounds__(256) void rotary_k(float* __restrict__ q,
                                                float* __restrict__ k)
{
    int gid = blockIdx.x * 256 + threadIdx.x;   // B*S*HEADS*16 threads
    int t   = gid & 15;
    int hh  = (gid >> 4) & 7;
    int row = gid >> 7;
    int s   = row & 1023;
    float inv = __powf(10000.f, -(float)(2 * t) / 32.f);
    float fr  = (float)s * inv;
    float c, sn; __sincosf(fr, &sn, &c);
    size_t base = (size_t)row * NDIM + hh * NDH;
    float q1 = q[base + t], q2 = q[base + t + 16];
    q[base + t]      = q1 * c - q2 * sn;
    q[base + t + 16] = q2 * c + q1 * sn;
    float k1 = k[base + t], k2 = k[base + t + 16];
    k[base + t]      = k1 * c - k2 * sn;
    k[base + t + 16] = k2 * c + k1 * sn;
}

// ---------------------------------------------------------------------------
// Flash attention: one wave per (b, h, 16-query block). Causal + relpos bias.
// Q*K^T and P*V on WMMA bf16; operand tiles in fragment order; staging is
// batched (4 loads issued together, then 4 packed b64 LDS stores).
// ---------------------------------------------------------------------------
__global__ __launch_bounds__(32) void flash_attn_k(
    const float* __restrict__ q, const float* __restrict__ k,
    const float* __restrict__ v, const float* __restrict__ relpos,
    float* __restrict__ o)
{
    int blk = blockIdx.x;
    int qb = blk & 63;
    int h  = (blk >> 6) & 7;
    int b  = blk >> 9;
    int lane = threadIdx.x;
    int half = lane >> 4, l16 = lane & 15;

    __shared__ __align__(16) __bf16 Qs2[16 * QSTRIDE];   // natural [q][d] order
    __shared__ __align__(16) __bf16 Ks [32 * KSTRIDE];   // [key][d] (B-frag natural)
    __shared__ __align__(16) __bf16 Vt [64 * BSTRIDE];   // transposed: [d][key]
    __shared__ __align__(16) __bf16 Ps2[16 * ASTRIDE];   // P in permuted A-order

    const size_t head_base = (size_t)b * NS * NDIM + (size_t)h * NDH;
    const int q0 = qb * 16;

    // stage Q (fold softmax scale 1/8 into Q), natural [q][d] order
    {
        float4 qr[8];
        #pragma unroll
        for (int j = 0; j < 8; ++j) {
            int i = lane + j * 32;                   // 256 float4 tasks
            int r = i >> 4, c4 = (i & 15) * 4;
            qr[j] = *(const float4*)(q + head_base + (size_t)(q0 + r) * NDIM + c4);
        }
        #pragma unroll
        for (int j = 0; j < 8; ++j) {
            int i = lane + j * 32;
            int r = i >> 4, c4 = (i & 15) * 4;
            int p = c4;
            st4(&Qs2[r * QSTRIDE + p], qr[j].x * 0.125f, qr[j].y * 0.125f,
                qr[j].z * 0.125f, qr[j].w * 0.125f);
        }
    }
    __syncthreads();

    bf16x16 qa0 = ld_frag(&Qs2[l16 * QSTRIDE + half * 16]);
    bf16x16 qa1 = ld_frag(&Qs2[l16 * QSTRIDE + 32 + half * 16]);

    f32x8 o0 = {}, o1 = {}, o2 = {}, o3 = {};
    float mrow[8], lrow[8];
    #pragma unroll
    for (int r = 0; r < 8; ++r) { mrow[r] = -3.0e38f; lrow[r] = 0.f; }

    const int kend = q0 + 16;
    for (int kb = 0; kb < kend; kb += 32) {
        // stage K tile 32x64 ([key][d], identity), batches of 4 float4
        #pragma unroll
        for (int cch = 0; cch < 4; ++cch) {
            float4 kr[4];
            #pragma unroll
            for (int j = 0; j < 4; ++j) {
                int i = lane + (cch * 4 + j) * 32;
                int r = i >> 4, c4 = (i & 15) * 4;
                kr[j] = *(const float4*)(k + head_base + (size_t)(kb + r) * NDIM + c4);
            }
            #pragma unroll
            for (int j = 0; j < 4; ++j) {
                int i = lane + (cch * 4 + j) * 32;
                int r = i >> 4, c4 = (i & 15) * 4;
                st4(&Ks[r * KSTRIDE + c4], kr[j].x, kr[j].y, kr[j].z, kr[j].w);
            }
        }
        // stage V transposed ([d][key]): 4 keys per fixed d, batches of 4
        #pragma unroll
        for (int cch = 0; cch < 4; ++cch) {
            float vr[4][4];
            #pragma unroll
            for (int j = 0; j < 4; ++j) {
                int i = lane + (cch * 4 + j) * 32;   // 512 tasks: d x r4-group
                int d  = i & 63;
                int r4 = (i >> 6) << 2;
                const float* vp = v + head_base + (size_t)(kb + r4) * NDIM + d;
                vr[j][0] = vp[0];        vr[j][1] = vp[NDIM];
                vr[j][2] = vp[2 * NDIM]; vr[j][3] = vp[3 * NDIM];
            }
            #pragma unroll
            for (int j = 0; j < 4; ++j) {
                int i = lane + (cch * 4 + j) * 32;
                int d  = i & 63;
                int r4 = (i >> 6) << 2;
                st4(&Vt[d * BSTRIDE + apos(r4)], vr[j][0], vr[j][1], vr[j][2], vr[j][3]);
            }
        }
        __syncthreads();

        // scores: S[16q x 32k] via 4 WMMAs (two key sub-tiles x two d-chunks)
        bf16x16 kb00 = ld_frag(&Ks[(l16) * KSTRIDE + half * 16]);
        bf16x16 kb01 = ld_frag(&Ks[(l16) * KSTRIDE + 32 + half * 16]);
        bf16x16 kb10 = ld_frag(&Ks[(16 + l16) * KSTRIDE + half * 16]);
        bf16x16 kb11 = ld_frag(&Ks[(16 + l16) * KSTRIDE + 32 + half * 16]);
        f32x8 s0 = {}, s1 = {};
        s0 = WMMA_BF16(qa0, kb00, s0);
        s0 = WMMA_BF16(qa1, kb01, s0);
        s1 = WMMA_BF16(qa0, kb10, s1);
        s1 = WMMA_BF16(qa1, kb11, s1);

        // bias + causal mask + online softmax (row = r+8*half, col key = l16)
        const int halfb = (l16 >> 3) & 1;
        #pragma unroll
        for (int r = 0; r < 8; ++r) {
            int qi = q0 + half * 8 + r;
            float sc0, sc1;
            #pragma unroll
            for (int t = 0; t < 2; ++t) {
                int kj = kb + t * 16 + l16;
                float val = (t == 0) ? s0[r] : s1[r];
                if (kj <= qi) {
                    int n = qi - kj;
                    int bucket;
                    if (n < 4) bucket = n;
                    else {
                        int vl = 4 + (int)(__logf((float)n * 0.25f) * 1.92359339f);
                        bucket = vl < 7 ? vl : 7;
                    }
                    val += relpos[bucket * NHEADS + h] * 8.f;
                } else {
                    val = -3.0e38f;
                }
                if (t == 0) sc0 = val; else sc1 = val;
            }
            float bm = fmaxf(sc0, sc1);
            #pragma unroll
            for (int off = 1; off < 16; off <<= 1) bm = fmaxf(bm, __shfl_xor(bm, off, 32));
            float mnew = fmaxf(mrow[r], bm);
            float corr = __expf(mrow[r] - mnew);
            float p0 = __expf(sc0 - mnew);
            float p1 = __expf(sc1 - mnew);
            float ps = p0 + p1;
            #pragma unroll
            for (int off = 1; off < 16; off <<= 1) ps += __shfl_xor(ps, off, 32);
            lrow[r] = lrow[r] * corr + ps;
            mrow[r] = mnew;
            o0[r] *= corr; o1[r] *= corr; o2[r] *= corr; o3[r] *= corr;
            // write P directly in permuted A-fragment order
            Ps2[(half * 8 + r) * ASTRIDE + halfb * 16 + (l16 & 7)]     = (__bf16)p0;
            Ps2[(half * 8 + r) * ASTRIDE + halfb * 16 + 8 + (l16 & 7)] = (__bf16)p1;
        }
        __syncthreads();

        // O += P[16x32] @ V[32x64]
        bf16x16 pa  = ld_frag(&Ps2[l16 * ASTRIDE + half * 16]);
        bf16x16 vb0 = ld_frag(&Vt[( 0 + l16) * BSTRIDE + half * 16]);
        bf16x16 vb1 = ld_frag(&Vt[(16 + l16) * BSTRIDE + half * 16]);
        bf16x16 vb2 = ld_frag(&Vt[(32 + l16) * BSTRIDE + half * 16]);
        bf16x16 vb3 = ld_frag(&Vt[(48 + l16) * BSTRIDE + half * 16]);
        o0 = WMMA_BF16(pa, vb0, o0);
        o1 = WMMA_BF16(pa, vb1, o1);
        o2 = WMMA_BF16(pa, vb2, o2);
        o3 = WMMA_BF16(pa, vb3, o3);
        __syncthreads();
    }

    // epilogue: normalize and write O
    #pragma unroll
    for (int r = 0; r < 8; ++r) {
        float invl = 1.f / fmaxf(lrow[r], 1e-20f);
        size_t rowbase = head_base + (size_t)(q0 + half * 8 + r) * NDIM;
        o[rowbase +  0 + l16] = o0[r] * invl;
        o[rowbase + 16 + l16] = o1[r] * invl;
        o[rowbase + 32 + l16] = o2[r] * invl;
        o[rowbase + 48 + l16] = o3[r] * invl;
    }
}

// ---------------------------------------------------------------------------
// Final LayerNorm
// ---------------------------------------------------------------------------
__global__ __launch_bounds__(256) void layernorm_k(
    const float* __restrict__ x, float* __restrict__ y,
    const float* __restrict__ g, const float* __restrict__ bsh)
{
    int row = blockIdx.x;
    const float* xr = x + (size_t)row * NDIM;
    __shared__ float red[16];
    float s = 0.f, s2 = 0.f;
    for (int i = threadIdx.x; i < NDIM; i += 256) {
        float v = xr[i]; s += v; s2 += v * v;
    }
    #pragma unroll
    for (int o = 16; o > 0; o >>= 1) { s += __shfl_xor(s, o, 32); s2 += __shfl_xor(s2, o, 32); }
    if ((threadIdx.x & 31) == 0) { red[threadIdx.x >> 5] = s; red[8 + (threadIdx.x >> 5)] = s2; }
    __syncthreads();
    if (threadIdx.x == 0) {
        float a = 0.f, bb = 0.f;
        #pragma unroll
        for (int i = 0; i < 8; ++i) { a += red[i]; bb += red[8 + i]; }
        red[0] = a; red[8] = bb;
    }
    __syncthreads();
    float mu  = red[0] * (1.f / NDIM);
    float var = red[8] * (1.f / NDIM) - mu * mu;
    float inv = rsqrtf(var + 1e-5f);
    for (int i = threadIdx.x; i < NDIM; i += 256)
        y[(size_t)row * NDIM + i] = (xr[i] - mu) * inv * g[i] + bsh[i];
}

// ---------------------------------------------------------------------------
// Output scatter: d_out (B,1024,1036) = concat(cls^T (B,NCLS,S), tail (B,S,12))
// ---------------------------------------------------------------------------
__global__ __launch_bounds__(256) void scatter_out_k(
    const float* __restrict__ clsout, const float* __restrict__ out524,
    float* __restrict__ dout)
{
    size_t gid = (size_t)blockIdx.x * 256 + threadIdx.x;
    const size_t total = (size_t)NB * 1024 * 1036;
    if (gid >= total) return;
    int j = (int)(gid % 1036);
    int i = (int)((gid / 1036) % 1024);
    int b = (int)(gid / (1036 * 1024));
    float val;
    if (j < 1024)
        val = clsout[((size_t)b * NS + j) * NNCLS + i];   // emb_g[b,i,j] = cls[b, s=j, n=i]
    else
        val = out524[((size_t)b * NS + i) * NDIN + NEMB + (j - 1024)];
    dout[gid] = val;
}

// ---------------------------------------------------------------------------
extern "C" void kernel_launch(void* const* d_in, const int* in_sizes, int n_in,
                              void* d_out, int out_size, void* d_ws, size_t ws_size,
                              hipStream_t stream) {
    const float* saml       = (const float*)d_in[0];
    const float* emb_table  = (const float*)d_in[1];
    const float* proj_in_w  = (const float*)d_in[2];
    const float* proj_in_b  = (const float*)d_in[3];
    const float* g_attn     = (const float*)d_in[4];
    const float* g_ff       = (const float*)d_in[5];
    const float* Wq         = (const float*)d_in[6];
    const float* Wk         = (const float*)d_in[7];
    const float* Wv         = (const float*)d_in[8];
    const float* Wo         = (const float*)d_in[9];
    const float* bo         = (const float*)d_in[10];
    const float* relpos     = (const float*)d_in[11];
    const float* W1         = (const float*)d_in[12];
    const float* b1         = (const float*)d_in[13];
    const float* W2         = (const float*)d_in[14];
    const float* b2         = (const float*)d_in[15];
    const float* ln_g       = (const float*)d_in[16];
    const float* ln_b       = (const float*)d_in[17];
    const float* proj_out_w = (const float*)d_in[18];
    const float* proj_out_b = (const float*)d_in[19];
    const float* cls_w      = (const float*)d_in[20];
    const float* cls_b      = (const float*)d_in[21];

    const int M = NB * NS;                       // 8192 rows
    float* ws = (float*)d_ws;
    const size_t ROWS = (size_t)M * NDIM;        // 4,194,304 floats
    float* h   = ws;
    float* hn  = h   + ROWS;
    float* qb  = hn  + ROWS;
    float* kb  = qb  + ROWS;
    float* vb  = kb  + ROWS;
    float* ff  = vb  + ROWS;                     // 16,777,216 floats
    float* out524 = qb;                          // reuse (final stage)
    float* clsout = ff;                          // reuse (final stage)

    auto gemm = [&](const float* A, const float* W, const float* bias,
                    const float* res, float* C, int N, int K, int lda, int act) {
        dim3 grid(M / TM, (N + TN - 1) / TN);
        gemm_bf16_wmma<<<grid, dim3(256), 0, stream>>>(A, W, bias, res, C, N, K, lda, act);
    };

    // 1) embedding + proj_in
    embed_proj_in_k<<<M, 256, 0, stream>>>(saml, emb_table, proj_in_w, proj_in_b, h);

    // 2) transformer layers
    for (int l = 0; l < NDEPTH; ++l) {
        const size_t wq_off = (size_t)l * NDIM * NDIM;
        scalenorm_k<<<M, 256, 0, stream>>>(h, hn, g_attn, l);
        gemm(hn, Wq + wq_off, nullptr, nullptr, qb, NDIM, NDIM, NDIM, 0);
        gemm(hn, Wk + wq_off, nullptr, nullptr, kb, NDIM, NDIM, NDIM, 0);
        gemm(hn, Wv + wq_off, nullptr, nullptr, vb, NDIM, NDIM, NDIM, 0);
        rotary_k<<<(M * NHEADS * 16) / 256, 256, 0, stream>>>(qb, kb);
        flash_attn_k<<<NB * NHEADS * (NS / 16), 32, 0, stream>>>(qb, kb, vb, relpos, hn);
        gemm(hn, Wo + wq_off, bo + (size_t)l * NDIM, h, h, NDIM, NDIM, NDIM, 0);
        scalenorm_k<<<M, 256, 0, stream>>>(h, hn, g_ff, l);
        gemm(hn, W1 + (size_t)l * NDIM * NFF, b1 + (size_t)l * NFF, nullptr, ff,
             NFF, NDIM, NDIM, 1 /* gelu */);
        gemm(ff, W2 + (size_t)l * NFF * NDIM, b2 + (size_t)l * NDIM, h, h,
             NDIM, NFF, NFF, 0);
    }

    // 3) final layernorm, proj_out, classifier head, scatter
    layernorm_k<<<M, 256, 0, stream>>>(h, hn, ln_g, ln_b);
    gemm(hn, proj_out_w, proj_out_b, nullptr, out524, NDIN, NDIM, NDIM, 0);
    gemm(out524, cls_w, cls_b, nullptr, clsout, NNCLS, NEMB, NDIN, 0);

    const size_t total = (size_t)NB * 1024 * 1036;
    scatter_out_k<<<(int)((total + 255) / 256), 256, 0, stream>>>(clsout, out524, (float*)d_out);

    (void)in_sizes; (void)n_in; (void)out_size; (void)ws_size;
}
